// v4d_2430951490187
// MI455X (gfx1250) — matrix operand order repaired and checked by reading, not compiled, not run on hardware
//
#include <hip/hip_runtime.h>
#include <hip/hip_bf16.h>

#define G      128
#define CD     12
#define NRAYS  4096
#define NSAMP  256
#define MLP_W  128
#define TM     64            // samples per block tile

#define NEARF      0.2f
#define FARF       3.4f
#define INTERVAL   ((FARF - NEARF) / (float)NSAMP)          // 0.0125
#define ACT_SHIFT  (-4.5951199f)                            // log(1/(1-0.01)-1)

typedef __attribute__((ext_vector_type(16))) _Float16     v16h;
typedef __attribute__((ext_vector_type(8)))  float        v8f;
typedef __attribute__((ext_vector_type(4)))  unsigned int u32x4;
typedef __attribute__((ext_vector_type(4)))  int          i32x4;
typedef __attribute__((ext_vector_type(8)))  int          i32x8;

// ---------------------------------------------------------------------------
// WMMA fragment helpers (layouts per CDNA5 ISA 7.12.2, wave32)
// ---------------------------------------------------------------------------

// A fragment: 16x32 f16 tile at (m0,k0) of row-major S[lda].
__device__ __forceinline__ v16h frag_a(const _Float16* S, int lda, int m0, int k0, int lane) {
  const int row = m0 + (lane & 15);
  const int kh  = (lane >> 4) << 3;            // 0 or 8
  const _Float16* p0 = S + row * lda + k0 + kh;
  const _Float16* p1 = p0 + 16;
  v16h a;
#pragma unroll
  for (int i = 0; i < 8; ++i) { a[i] = p0[i]; a[8 + i] = p1[i]; }
  return a;
}

// B fragment: 32x16 f16 tile at (k0,n0) of row-major W[ldb].
__device__ __forceinline__ v16h frag_b(const _Float16* W, int ldb, int k0, int n0, int lane) {
  const int kh = (lane >> 4) << 3;            // 0 or 8
  const _Float16* p = W + (k0 + kh) * ldb + n0 + (lane & 15);
  v16h b;
#pragma unroll
  for (int i = 0; i < 8; ++i) { b[i] = p[i * ldb]; b[8 + i] = p[(16 + i) * ldb]; }
  return b;
}

__device__ __forceinline__ v8f wmma_f16(v16h a, v16h b, v8f c) {
  return __builtin_amdgcn_wmma_f32_16x16x32_f16(false, a, false, b, (short)0, c, false, false);
}

__device__ __forceinline__ void store_relu(v8f c, _Float16* H, int ldh, int m0, int n0,
                                           int lane, const float* bias) {
  const int col   = n0 + (lane & 15);
  const float bv  = bias[col];
  const int rbase = m0 + ((lane >> 4) << 3);
#pragma unroll
  for (int r = 0; r < 8; ++r) {
    float v = c[r] + bv;
    H[(rbase + r) * ldh + col] = (_Float16)(v > 0.f ? v : 0.f);
  }
}

// ---------------------------------------------------------------------------
// Tensor Data Mover: stage a 2D fp16 matrix (rows x dwords_per_row dwords)
// from global into LDS. D# packing per CDNA5 ISA ch.8 (group0 + group1).
// ---------------------------------------------------------------------------
__device__ __forceinline__ void tdm_load_2d(unsigned lds_off, const void* gptr,
                                            unsigned rows, unsigned dpr /*dwords per row*/) {
  const unsigned long long ga = (unsigned long long)gptr;
  u32x4 g0;
  g0[0] = 1u;                                           // count=1, user descriptor
  g0[1] = lds_off;                                      // lds_addr
  g0[2] = (unsigned)(ga & 0xffffffffu);                 // global_addr[31:0]
  g0[3] = (unsigned)((ga >> 32) & 0x01ffffffu) | (2u << 30);  // addr[56:32] | type=2
  i32x8 g1;
  g1[0] = (int)(2u << 16);                              // data_size = 2 (4-byte elems)
  g1[1] = (int)((dpr & 0xffffu) << 16);                 // tensor_dim0[15:0]
  g1[2] = (int)((dpr >> 16) | ((rows & 0xffffu) << 16));// dim0[31:16] | dim1[15:0]
  g1[3] = (int)((rows >> 16) | (dpr << 16));            // dim1[31:16] | tile_dim0
  g1[4] = (int)(rows & 0xffffu);                        // tile_dim1 (tile_dim2 = 0)
  g1[5] = (int)dpr;                                     // tensor_dim0_stride[31:0]
  g1[6] = 0;
  g1[7] = 0;
  i32x4 z4 = {0, 0, 0, 0};
  i32x8 z8 = {0, 0, 0, 0, 0, 0, 0, 0};
#if __clang_major__ >= 23
  __builtin_amdgcn_tensor_load_to_lds(g0, g1, z4, z4, z8, 0);   // amdgpu-toolchain
#else
  __builtin_amdgcn_tensor_load_to_lds(g0, g1, z4, z4, 0);       // ROCm 7.x
#endif
}

// ---------------------------------------------------------------------------
// Per-sample geometry, and half-width trilinear gather (6 channels)
// ---------------------------------------------------------------------------
struct SamplePos { float x01, y01, z01, dx, dy, dz, tt; };

__device__ __forceinline__ SamplePos sample_pos(const float* __restrict__ ro,
                                                const float* __restrict__ rd,
                                                const float* __restrict__ times, int p) {
  SamplePos sp;
  const int n = p >> 8, s = p & 255;
  float dx = rd[n * 3 + 0], dy = rd[n * 3 + 1], dz = rd[n * 3 + 2];
  const float inv = rsqrtf(dx * dx + dy * dy + dz * dz);
  sp.dx = dx * inv; sp.dy = dy * inv; sp.dz = dz * inv;
  const float t = NEARF + (FARF - NEARF) * (float)s / (float)(NSAMP - 1);
  sp.x01 = fminf(fmaxf((ro[n * 3 + 0] + sp.dx * t + 1.f) * 0.5f, 0.f), 1.f);
  sp.y01 = fminf(fmaxf((ro[n * 3 + 1] + sp.dy * t + 1.f) * 0.5f, 0.f), 1.f);
  sp.z01 = fminf(fmaxf((ro[n * 3 + 2] + sp.dz * t + 1.f) * 0.5f, 0.f), 1.f);
  sp.tt  = times[n];
  return sp;
}

__device__ __forceinline__ void trilerp6(const float* __restrict__ g,
                                         float x01, float y01, float z01,
                                         int c0, float* __restrict__ out) {
  const float px = x01 * (float)(G - 1);
  const float py = y01 * (float)(G - 1);
  const float pz = z01 * (float)(G - 1);
  int ix = (int)floorf(px); ix = ix < 0 ? 0 : (ix > G - 2 ? G - 2 : ix);
  int iy = (int)floorf(py); iy = iy < 0 ? 0 : (iy > G - 2 ? G - 2 : iy);
  int iz = (int)floorf(pz); iz = iz < 0 ? 0 : (iz > G - 2 ? G - 2 : iz);
  const float fx = fminf(fmaxf(px - (float)ix, 0.f), 1.f);
  const float fy = fminf(fmaxf(py - (float)iy, 0.f), 1.f);
  const float fz = fminf(fmaxf(pz - (float)iz, 0.f), 1.f);
  const int SX = G * G * CD, SY = G * CD, SZ = CD;
  const int base = ((ix * G + iy) * G + iz) * CD + c0;
  const float w000 = (1 - fx) * (1 - fy) * (1 - fz);
  const float w001 = (1 - fx) * (1 - fy) * fz;
  const float w010 = (1 - fx) * fy * (1 - fz);
  const float w011 = (1 - fx) * fy * fz;
  const float w100 = fx * (1 - fy) * (1 - fz);
  const float w101 = fx * (1 - fy) * fz;
  const float w110 = fx * fy * (1 - fz);
  const float w111 = fx * fy * fz;
#pragma unroll
  for (int c = 0; c < 6; ++c) {
    out[c] = w000 * g[base + c]           + w001 * g[base + SZ + c]
           + w010 * g[base + SY + c]      + w011 * g[base + SY + SZ + c]
           + w100 * g[base + SX + c]      + w101 * g[base + SX + SZ + c]
           + w110 * g[base + SX + SY + c] + w111 * g[base + SX + SY + SZ + c];
  }
}

// ---------------------------------------------------------------------------
// Kernel 0: weights -> fp16 (zero-padded to WMMA-friendly shapes)
// ---------------------------------------------------------------------------
__global__ void prep_weights(const float* ow1, const float* ow2,
                             const float* rw1, const float* rw2, const float* rw3,
                             _Float16* ow1h, _Float16* ow2h,
                             _Float16* rw1h, _Float16* rw2h, _Float16* rw3h) {
  const int i = blockIdx.x * blockDim.x + threadIdx.x;
  const int stride = gridDim.x * blockDim.x;
  for (int idx = i; idx < 64 * 128; idx += stride) {
    int k = idx >> 7, n = idx & 127;
    ow1h[idx] = (_Float16)(k < 56 ? ow1[k * 128 + n] : 0.f);
  }
  for (int idx = i; idx < 128 * 128; idx += stride) ow2h[idx] = (_Float16)ow2[idx];
  for (int idx = i; idx < 96 * 128; idx += stride) {
    int k = idx >> 7, n = idx & 127;
    rw1h[idx] = (_Float16)(k < 83 ? rw1[k * 128 + n] : 0.f);
  }
  for (int idx = i; idx < 128 * 128; idx += stride) rw2h[idx] = (_Float16)rw2[idx];
  for (int idx = i; idx < 128 * 16; idx += stride) {
    int k = idx >> 4, n = idx & 15;
    rw3h[idx] = (_Float16)(n < 3 ? rw3[k * 3 + n] : 0.f);
  }
}

// ---------------------------------------------------------------------------
// Kernel 1: fused featurize + occupancy MLP (WMMA + TDM) -> raw density
// LDS: Xs 8KB + H1 16KB + Alias 16KB (W1 during GEMM1, H2 during GEMM2/L3)
// ---------------------------------------------------------------------------
__global__ __launch_bounds__(256) void occ_kernel(
    const float* __restrict__ rays_o, const float* __restrict__ rays_d,
    const float* __restrict__ times,  const float* __restrict__ dgrid,
    const _Float16* __restrict__ w1h, const float* __restrict__ b1,
    const _Float16* __restrict__ w2h, const float* __restrict__ b2,
    const float* __restrict__ w3,     const float* __restrict__ b3,
    float* __restrict__ raw) {
  __shared__ _Float16 Xs[TM * 64];
  __shared__ _Float16 H1[TM * MLP_W];
  __shared__ __align__(16) _Float16 Alias[TM * MLP_W];   // W1 (64x128) then H2 (64x128)
  _Float16* W1s = Alias;
  _Float16* H2  = Alias;

  const int tid = threadIdx.x;
  const int p0  = blockIdx.x * TM;
  const int wv  = tid >> 5, lane = tid & 31;

  // async: TDM pulls layer-1 weights into LDS while we featurize
  if (wv == 0)
    tdm_load_2d((unsigned)(size_t)(void*)Alias, (const void*)w1h, 64, 64);
  __builtin_prefetch(w2h, 0, 0);

  // featurize: 4 threads per sample
  {
    const int sm = tid & (TM - 1);
    const int q  = tid >> 6;                 // 0..3
    const SamplePos sp = sample_pos(rays_o, rays_d, times, p0 + sm);
    _Float16* xr = &Xs[sm * 64];
    if (q < 2) {                             // trilinear, 6 channels each
      float f[6];
      trilerp6(dgrid, sp.x01, sp.y01, sp.z01, q * 6, f);
#pragma unroll
      for (int c = 0; c < 6; ++c) xr[q * 6 + c] = (_Float16)f[c];
    } else if (q == 2) {                     // xyzt + sin block
      const float vt[4] = {sp.x01, sp.y01, sp.z01, sp.tt};
#pragma unroll
      for (int d = 0; d < 4; ++d) {
        xr[12 + d] = (_Float16)vt[d];
#pragma unroll
        for (int f = 0; f < 5; ++f)
          xr[16 + d * 5 + f] = (_Float16)__sinf(vt[d] * (float)(1 << f));
      }
    } else {                                 // cos block + zero pad
      const float vt[4] = {sp.x01, sp.y01, sp.z01, sp.tt};
#pragma unroll
      for (int d = 0; d < 4; ++d)
#pragma unroll
        for (int f = 0; f < 5; ++f)
          xr[36 + d * 5 + f] = (_Float16)__cosf(vt[d] * (float)(1 << f));
#pragma unroll
      for (int c = 56; c < 64; ++c) xr[c] = (_Float16)0.f;
    }
  }
  if (wv == 0) __builtin_amdgcn_s_wait_tensorcnt(0);
  __syncthreads();

  // layer 1: H1[64x128] = relu(Xs[64x64] @ W1s[64x128] + b1)   (A,B from LDS)
#pragma unroll
  for (int mt = 0; mt < 4; ++mt) {
    v8f acc = {};
#pragma unroll
    for (int k0 = 0; k0 < 64; k0 += 32)
      acc = wmma_f16(frag_a(Xs, 64, mt * 16, k0, lane),
                     frag_b(W1s, 128, k0, wv * 16, lane), acc);
    store_relu(acc, H1, MLP_W, mt * 16, wv * 16, lane, b1);
  }
  __syncthreads();

  // layer 2: H2 = relu(H1 @ W2 + b2), K=128 (B from global, L2-resident)
#pragma unroll
  for (int mt = 0; mt < 4; ++mt) {
    v8f acc = {};
#pragma unroll
    for (int k0 = 0; k0 < 128; k0 += 32)
      acc = wmma_f16(frag_a(H1, MLP_W, mt * 16, k0, lane),
                     frag_b(w2h, 128, k0, wv * 16, lane), acc);
    store_relu(acc, H2, MLP_W, mt * 16, wv * 16, lane, b2);
  }
  __syncthreads();

  // layer 3: 1-wide output, 4 threads per sample + shuffle reduce
  {
    const int srow = tid >> 2;               // 0..63
    const int q    = tid & 3;
    const _Float16* h = &H2[srow * MLP_W + q * 32];
    float sum = 0.f;
#pragma unroll 8
    for (int k = 0; k < 32; ++k) sum += (float)h[k] * w3[q * 32 + k];
    sum += __shfl_xor(sum, 1, 32);
    sum += __shfl_xor(sum, 2, 32);
    if (q == 0) raw[p0 + srow] = sum + b3[0];
  }
}

// ---------------------------------------------------------------------------
// Kernel 2: per-ray alpha compositing scan -> weights + background into d_out
// ---------------------------------------------------------------------------
__global__ __launch_bounds__(256) void composite_kernel(const float* __restrict__ raw,
                                                        float* __restrict__ wbuf,
                                                        float* __restrict__ out) {
  const int n = blockIdx.x * blockDim.x + threadIdx.x;
  if (n >= NRAYS) return;
  float cum = 1.f;
  for (int s = 0; s < NSAMP; ++s) {
    const float x  = raw[n * NSAMP + s] + ACT_SHIFT;
    const float sp = (x > 20.f) ? x : log1pf(__expf(x));
    const float alpha = 1.f - __expf(-sp * INTERVAL);
    wbuf[n * NSAMP + s] = alpha * cum;
    float om = 1.f - alpha;
    if (om < 1e-10f) om = 1e-10f;
    cum *= om;
  }
  out[n * 3 + 0] = cum;
  out[n * 3 + 1] = cum;
  out[n * 3 + 2] = cum;
}

// ---------------------------------------------------------------------------
// Kernel 3: fused featurize + RGB MLP (all-WMMA + TDM) + weighted reduction
// LDS: Xs 12KB + H1 16KB + Alias 24KB (W1 96x128, later H2 64x128)
// ---------------------------------------------------------------------------
__global__ __launch_bounds__(256) void rgb_kernel(
    const float* __restrict__ rays_o, const float* __restrict__ rays_d,
    const float* __restrict__ times,  const float* __restrict__ kgrid,
    const _Float16* __restrict__ w1h, const float* __restrict__ b1,
    const _Float16* __restrict__ w2h, const float* __restrict__ b2,
    const _Float16* __restrict__ w3h, const float* __restrict__ b3,
    const float* __restrict__ wbuf,   float* __restrict__ out) {
  __shared__ _Float16 Xs[TM * 96];
  __shared__ _Float16 H1[TM * MLP_W];
  __shared__ __align__(16) _Float16 Alias[96 * MLP_W];   // W1 (96x128) then H2 (64x128)
  _Float16* W1s = Alias;
  _Float16* H2  = Alias;

  const int tid = threadIdx.x;
  const int p0  = blockIdx.x * TM;
  const int ray = p0 >> 8;
  const int wv  = tid >> 5, lane = tid & 31;

  if (wv == 0)
    tdm_load_2d((unsigned)(size_t)(void*)Alias, (const void*)w1h, 96, 64);
  __builtin_prefetch(w2h, 0, 0);

  // featurize: 4 threads per sample
  {
    const int sm = tid & (TM - 1);
    const int q  = tid >> 6;
    const SamplePos sp = sample_pos(rays_o, rays_d, times, p0 + sm);
    _Float16* xr = &Xs[sm * 96];
    if (q < 2) {                             // k0 trilinear, 6 channels each
      float f[6];
      trilerp6(kgrid, sp.x01, sp.y01, sp.z01, q * 6, f);
#pragma unroll
      for (int c = 0; c < 6; ++c) xr[71 + q * 6 + c] = (_Float16)f[c];
      if (q == 1)
#pragma unroll
        for (int c = 83; c < 96; ++c) xr[c] = (_Float16)0.f;
    } else if (q == 2) {                     // xyz positional encoding
      const float vp[3] = {sp.x01, sp.y01, sp.z01};
#pragma unroll
      for (int d = 0; d < 3; ++d) {
        xr[d] = (_Float16)vp[d];
#pragma unroll
        for (int f = 0; f < 5; ++f) {
          const float ang = vp[d] * (float)(1 << f);
          xr[3  + d * 5 + f] = (_Float16)__sinf(ang);
          xr[18 + d * 5 + f] = (_Float16)__cosf(ang);
        }
      }
    } else {                                 // view + time encodings
      const float vd[3] = {sp.dx, sp.dy, sp.dz};
#pragma unroll
      for (int d = 0; d < 3; ++d) {
        xr[33 + d] = (_Float16)vd[d];
#pragma unroll
        for (int f = 0; f < 4; ++f) {
          const float ang = vd[d] * (float)(1 << f);
          xr[36 + d * 4 + f] = (_Float16)__sinf(ang);
          xr[48 + d * 4 + f] = (_Float16)__cosf(ang);
        }
      }
      xr[60] = (_Float16)sp.tt;
#pragma unroll
      for (int f = 0; f < 5; ++f) {
        const float ang = sp.tt * (float)(1 << f);
        xr[61 + f] = (_Float16)__sinf(ang);
        xr[66 + f] = (_Float16)__cosf(ang);
      }
    }
  }
  if (wv == 0) __builtin_amdgcn_s_wait_tensorcnt(0);
  __syncthreads();

  // layer 1: K=96 (A,B from LDS)
#pragma unroll
  for (int mt = 0; mt < 4; ++mt) {
    v8f acc = {};
#pragma unroll
    for (int k0 = 0; k0 < 96; k0 += 32)
      acc = wmma_f16(frag_a(Xs, 96, mt * 16, k0, lane),
                     frag_b(W1s, 128, k0, wv * 16, lane), acc);
    store_relu(acc, H1, MLP_W, mt * 16, wv * 16, lane, b1);
  }
  __syncthreads();

  // layer 2: K=128
#pragma unroll
  for (int mt = 0; mt < 4; ++mt) {
    v8f acc = {};
#pragma unroll
    for (int k0 = 0; k0 < 128; k0 += 32)
      acc = wmma_f16(frag_a(H1, MLP_W, mt * 16, k0, lane),
                     frag_b(w2h, 128, k0, wv * 16, lane), acc);
    store_relu(acc, H2, MLP_W, mt * 16, wv * 16, lane, b2);
  }
  __syncthreads();

  // layer 3: WMMA with N padded 3->16; sigmoid + weighted reduce + atomicAdd
  if (wv < 4) {
    v8f acc = {};
#pragma unroll
    for (int k0 = 0; k0 < 128; k0 += 32)
      acc = wmma_f16(frag_a(H2, MLP_W, wv * 16, k0, lane),
                     frag_b(w3h, 16, k0, 0, lane), acc);
    const int col   = lane & 15;
    const float bv  = (col < 3) ? b3[col] : 0.f;
    const int rbase = wv * 16 + ((lane >> 4) << 3);
    float partial = 0.f;
#pragma unroll
    for (int r = 0; r < 8; ++r) {
      const int row = rbase + r;
      const float rgb = 1.f / (1.f + __expf(-(acc[r] + bv)));
      partial += rgb * wbuf[p0 + row];
    }
    partial += __shfl_xor(partial, 16, 32);
    if (lane < 16 && col < 3) atomicAdd(&out[ray * 3 + col], partial);
  }
}

// ---------------------------------------------------------------------------
// Host launcher
// ---------------------------------------------------------------------------
extern "C" void kernel_launch(void* const* d_in, const int* in_sizes, int n_in,
                              void* d_out, int out_size, void* d_ws, size_t ws_size,
                              hipStream_t stream) {
  (void)in_sizes; (void)n_in; (void)out_size; (void)ws_size;
  const float* rays_o = (const float*)d_in[0];
  const float* rays_d = (const float*)d_in[1];
  const float* times  = (const float*)d_in[2];
  const float* dgrid  = (const float*)d_in[3];
  const float* kgrid  = (const float*)d_in[4];
  const float* ow1 = (const float*)d_in[5];
  const float* ob1 = (const float*)d_in[6];
  const float* ow2 = (const float*)d_in[7];
  const float* ob2 = (const float*)d_in[8];
  const float* ow3 = (const float*)d_in[9];
  const float* ob3 = (const float*)d_in[10];
  const float* rw1 = (const float*)d_in[11];
  const float* rb1 = (const float*)d_in[12];
  const float* rw2 = (const float*)d_in[13];
  const float* rb2 = (const float*)d_in[14];
  const float* rw3 = (const float*)d_in[15];
  const float* rb3 = (const float*)d_in[16];

  char* ws = (char*)d_ws;
  size_t off = 0;
  auto take = [&](size_t bytes) -> char* {
    char* p = ws + off;
    off = (off + bytes + 255) & ~(size_t)255;
    return p;
  };
  _Float16* ow1h = (_Float16*)take(64  * 128 * sizeof(_Float16));
  _Float16* ow2h = (_Float16*)take(128 * 128 * sizeof(_Float16));
  _Float16* rw1h = (_Float16*)take(96  * 128 * sizeof(_Float16));
  _Float16* rw2h = (_Float16*)take(128 * 128 * sizeof(_Float16));
  _Float16* rw3h = (_Float16*)take(128 * 16  * sizeof(_Float16));
  float* raw  = (float*)take((size_t)NRAYS * NSAMP * sizeof(float));
  float* wbuf = (float*)take((size_t)NRAYS * NSAMP * sizeof(float));

  prep_weights<<<64, 256, 0, stream>>>(ow1, ow2, rw1, rw2, rw3,
                                       ow1h, ow2h, rw1h, rw2h, rw3h);

  const int nblocks = (NRAYS * NSAMP) / TM;   // 16384
  occ_kernel<<<nblocks, 256, 0, stream>>>(rays_o, rays_d, times, dgrid,
                                          ow1h, ob1, ow2h, ob2, ow3, ob3, raw);

  composite_kernel<<<(NRAYS + 255) / 256, 256, 0, stream>>>(raw, wbuf, (float*)d_out);

  rgb_kernel<<<nblocks, 256, 0, stream>>>(rays_o, rays_d, times, kgrid,
                                          rw1h, rb1, rw2h, rb2, rw3h, rb3,
                                          wbuf, (float*)d_out);
}
